// ImprovedGNNAutoEncoder_70866960384532
// MI455X (gfx1250) — matrix operand order repaired and checked by reading, not compiled, not run on hardware
//
#include <hip/hip_runtime.h>

// ---------------------------------------------------------------------------
// ImprovedGNNAutoEncoder for MI455X (gfx1250, wave32, WMMA bf16->f32)
//
// Roofline: edge aggregation dominates (~8 GB -> ~0.34 ms @ 23.3 TB/s);
// GEMMs are 46 GFLOP and trivial on v_wmma_f32_16x16x32_bf16.
//   * activations kept in bf16 (halves gather traffic)
//   * aggregation accumulates fp32 via global_atomic_add_f32, 2 feat/thread
//   * self-loop folded into aggregation init (no extra atomic pass)
//   * GEMM B panel staged in LDS (ds_load_b128 feeds WMMA), A from global
//     (A fits in the 192MB L2, so cross-block re-reads never hit HBM)
//   * decoder BN stats fused into the GEMM epilogue (saves 102MB/layer pass)
// ---------------------------------------------------------------------------

#define GN 100000     // nodes
#define GE 1600000    // directed edges
#define GIN 128       // in channels
#define GH 256        // hidden
#define GL 64         // latent
#define GEPS 1e-5f

typedef __attribute__((ext_vector_type(16))) __bf16 v16bf;
typedef __attribute__((ext_vector_type(8)))  float  v8f;

union FragBF { uint4 q[2]; v16bf v; };

__device__ __forceinline__ unsigned short f2bf(float f) {
  unsigned u = __float_as_uint(f);
  u += 0x7FFFu + ((u >> 16) & 1u);       // round-to-nearest-even
  return (unsigned short)(u >> 16);
}
__device__ __forceinline__ float bf2f(unsigned short h) {
  return __uint_as_float(((unsigned)h) << 16);
}

// ---------------------------------------------------------------- utility ---
__global__ void cvt_f32_bf16_k(const float* __restrict__ in,
                               unsigned short* __restrict__ out, size_t n) {
  size_t i = (size_t)blockIdx.x * blockDim.x + threadIdx.x;
  if (i < n) out[i] = f2bf(in[i]);
}

__global__ void zero_f32_k(float* __restrict__ p, int n) {
  int i = blockIdx.x * blockDim.x + threadIdx.x;
  if (i < n) p[i] = 0.0f;
}

// -------------------------------------------------------------- degree ------
__global__ void deg_init_k(float* __restrict__ deg) {
  int i = blockIdx.x * blockDim.x + threadIdx.x;
  if (i < GN) deg[i] = 1.0f;                 // self loop
}
__global__ void deg_acc_k(float* __restrict__ deg,
                          const long long* __restrict__ ei) {
  int e = blockIdx.x * blockDim.x + threadIdx.x;
  if (e < GE) atomicAdd(&deg[(int)ei[GE + e]], 1.0f);
}
__global__ void dinv_k(float* __restrict__ deg) {
  int i = blockIdx.x * blockDim.x + threadIdx.x;
  if (i < GN) deg[i] = rsqrtf(fmaxf(deg[i], 1.0f));   // in place -> dinv
}

// ------------------------------------------------- weight pack (B fragment) -
// Rearranges row-major f32 W[Kin x Kout] into bf16 fragments in the exact
// per-lane k order this GEMM uses for the WMMA B operand (32x16 bf16, contiguous k as for A): element e of lane l
// holds B[K = e + 16*(l>=16)][n = l%16].
__global__ void pack_w_k(const float* __restrict__ W,
                         unsigned short* __restrict__ Wp, int Kin, int Kout) {
  int i = blockIdx.x * blockDim.x + threadIdx.x;
  if (i >= Kin * Kout) return;
  int k = i / Kout, n = i % Kout;
  int kt = k >> 5, kr = k & 31;
  int nt = n >> 4, nr = n & 15;
  int lane = nr + ((kr >> 4) << 4);
  int e    = kr & 15;
  int ntiles = Kout >> 4;
  size_t dst = ((size_t)(kt * ntiles + nt) << 9) + (lane << 4) + e;
  Wp[dst] = f2bf(W[i]);
}

// ----------------------------------------------------------- WMMA GEMM ------
// C[M x Kout] = A_bf16[M x Kin] @ Wpacked_bf16 (+bias). One wave per 16x16 C
// tile, 8 waves per block share the blockIdx.y B panel staged in LDS.
// Optional fused per-column sum/sumsq (BatchNorm statistics).
__global__ __launch_bounds__(256) void gemm_wmma_k(
    const unsigned short* __restrict__ A,
    const unsigned short* __restrict__ Wp,
    float* __restrict__ outF32, unsigned short* __restrict__ outBF16,
    const float* __restrict__ bias, float* __restrict__ statsOut,
    int mTiles, int Kin, int Kout) {
  __shared__ unsigned short smem[4096];          // B panel: up to (256/32)*512

  int wave = threadIdx.x >> 5;
  int lane = threadIdx.x & 31;
  int tileRow = blockIdx.x * 8 + wave;
  bool active = tileRow < mTiles;
  int tr = active ? tileRow : (mTiles - 1);      // clamp: all waves hit barrier
  int m0 = tr << 4;
  int n0 = blockIdx.y << 4;
  int laneHi = lane >> 4;          // 0 | 1
  int mr = lane & 15;
  int ntiles = Kout >> 4;

  // Cooperative stage of this block's B panel (all K tiles for n-tile) in LDS.
  for (int idx = threadIdx.x * 16; idx < Kin * 16; idx += 256 * 16) {
    int kt = idx >> 9;
    int off = idx & 511;
    const uint4* src = (const uint4*)(
        Wp + (((size_t)kt * ntiles + blockIdx.y) << 9) + off);
    uint4* dst = (uint4*)(smem + idx);
    dst[0] = src[0];
    dst[1] = src[1];
  }
  __syncthreads();

  v8f acc = {};
  const unsigned short* Arow = A + (size_t)(m0 + mr) * Kin;
  for (int kk = 0; kk < Kin; kk += 32) {
    // A fragment 16x32: lane<16 -> K {kk..kk+15}; lane>=16 -> K {kk+16..kk+31}
    FragBF a;
    int kb = kk + laneHi * 16;
    a.q[0] = *(const uint4*)(Arow + kb);
    a.q[1] = *(const uint4*)(Arow + kb + 8);
    // B fragment: 32 contiguous bytes per lane from LDS (ds_load_b128 x2)
    FragBF b;
    const unsigned short* bp = smem + ((kk >> 5) << 9) + (lane << 4);
    b.q[0] = *(const uint4*)(bp);
    b.q[1] = *(const uint4*)(bp + 8);
    acc = __builtin_amdgcn_wmma_f32_16x16x32_bf16(
        false, a.v, false, b.v, (short)0, acc, false, false);
  }

  if (!active) return;                           // past the barrier: safe
  int col = n0 + mr;
  float bv = bias ? bias[col] : 0.0f;
  // C fragment: element r -> row m0 + r + 8*laneHi, column col
  if (outF32) {
    float s = 0.0f, s2 = 0.0f;
#pragma unroll
    for (int r = 0; r < 8; ++r) {
      float v = acc[r] + bv;
      outF32[(size_t)(m0 + r + laneHi * 8) * Kout + col] = v;
      s += v;
      s2 += v * v;
    }
    if (statsOut) {                              // fused BatchNorm statistics
      atomicAdd(&statsOut[col], s);
      atomicAdd(&statsOut[Kout + col], s2);
    }
  } else {
#pragma unroll
    for (int r = 0; r < 8; ++r)
      outBF16[(size_t)(m0 + r + laneHi * 8) * Kout + col] = f2bf(acc[r] + bv);
  }
}

// --------------------------------------------------------- aggregation ------
// agg[n][f] = bias[f] + dinv[n]^2 * h[n][f]   (self loop folded in)
__global__ void agg_init_k(float* __restrict__ agg,
                           const unsigned short* __restrict__ h,
                           const float* __restrict__ bias,
                           const float* __restrict__ dinv, int F) {
  size_t i = (size_t)blockIdx.x * blockDim.x + threadIdx.x;
  if (i >= (size_t)GN * F) return;
  size_t n = i / F;
  int f = (int)(i - n * F);
  float c = dinv[n];
  agg[i] = bias[f] + c * c * bf2f(h[i]);
}

// agg[dst][f] += dinv[src]*dinv[dst] * h[src][f]
// bf16 gather (packed 2 features/thread), fp32 atomic accumulate.
__global__ void agg_edges_k(float* __restrict__ agg,
                            const unsigned short* __restrict__ h,
                            const float* __restrict__ dinv,
                            const long long* __restrict__ ei, int log2Fh) {
  size_t i = (size_t)blockIdx.x * blockDim.x + threadIdx.x;  // over E * F/2
  size_t e = i >> log2Fh;
  if (e >= GE) return;
  int fp = (int)(i & ((1u << log2Fh) - 1u)) << 1;
  int F = 2 << log2Fh;
  int s = (int)ei[e];
  int d = (int)ei[GE + e];
  float coef = dinv[s] * dinv[d];
  unsigned pk = *(const unsigned*)(h + (size_t)s * F + fp);
  float* dstp = agg + (size_t)d * F + fp;
  atomicAdd(dstp,     coef * bf2f((unsigned short)(pk & 0xFFFFu)));
  atomicAdd(dstp + 1, coef * bf2f((unsigned short)(pk >> 16)));
}

// ----------------------------------------------------------- BatchNorm ------
// stats[f] = sum, stats[F+f] = sumsq   (blockDim.x == F == 256)
__global__ void bn_stats_k(const float* __restrict__ X,
                           float* __restrict__ stats, int rowsPerBlock) {
  int f = threadIdx.x;
  int r0 = blockIdx.x * rowsPerBlock;
  float s = 0.0f, s2 = 0.0f;
  for (int r = 0; r < rowsPerBlock; ++r) {
    int row = r0 + r;
    if (row >= GN) break;
    float v = X[(size_t)row * GH + f];
    s += v;
    s2 += v * v;
  }
  atomicAdd(&stats[f], s);
  atomicAdd(&stats[GH + f], s2);
}

// out_bf16 = relu(g*(x-mean)*rsqrt(var+eps)+b), feeds next bf16 GEMM
__global__ void bn_apply_k(const float* __restrict__ X,
                           const float* __restrict__ stats,
                           const float* __restrict__ g,
                           const float* __restrict__ be,
                           unsigned short* __restrict__ out) {
  size_t i = (size_t)blockIdx.x * blockDim.x + threadIdx.x;
  if (i >= (size_t)GN * GH) return;
  int f = (int)(i & (GH - 1));
  float inv_n = 1.0f / (float)GN;
  float mean = stats[f] * inv_n;
  float var = stats[GH + f] * inv_n - mean * mean;
  float sc = g[f] * rsqrtf(var + GEPS);
  float sh = be[f] - mean * sc;
  out[i] = f2bf(fmaxf(X[i] * sc + sh, 0.0f));
}

// z: write f32 to d_out slice + bf16 copy for decoder GEMM input
__global__ void z_out_k(const float* __restrict__ z, float* __restrict__ zo,
                        unsigned short* __restrict__ zb, size_t n) {
  size_t i = (size_t)blockIdx.x * blockDim.x + threadIdx.x;
  if (i >= n) return;
  float v = z[i];
  zo[i] = v;
  zb[i] = f2bf(v);
}

// ---------------------------------------------------------------------------
extern "C" void kernel_launch(void* const* d_in, const int* in_sizes, int n_in,
                              void* d_out, int out_size, void* d_ws,
                              size_t ws_size, hipStream_t stream) {
  (void)in_sizes; (void)n_in; (void)out_size; (void)ws_size;

  const float*     x   = (const float*)d_in[0];
  const long long* ei  = (const long long*)d_in[1];   // int64 edge_index
  const float* W1 = (const float*)d_in[2],  *b1 = (const float*)d_in[3];
  const float* g1 = (const float*)d_in[4],  *be1 = (const float*)d_in[5];
  const float* W2 = (const float*)d_in[6],  *b2 = (const float*)d_in[7];
  const float* g2 = (const float*)d_in[8],  *be2 = (const float*)d_in[9];
  const float* W3 = (const float*)d_in[10], *b3 = (const float*)d_in[11];
  const float* dW1 = (const float*)d_in[12], *db1 = (const float*)d_in[13];
  const float* dg1 = (const float*)d_in[14], *dbe1 = (const float*)d_in[15];
  const float* dW2 = (const float*)d_in[16], *db2 = (const float*)d_in[17];
  const float* dg2 = (const float*)d_in[18], *dbe2 = (const float*)d_in[19];
  const float* dW3 = (const float*)d_in[20], *db3 = (const float*)d_in[21];

  float* x_hat = (float*)d_out;                        // [N, IN]
  float* z_f32 = (float*)d_out + (size_t)GN * GIN;     // [N, L]

  // ---- workspace layout (256B aligned offsets) ----
  char* ws = (char*)d_ws;
  float*          dinv  = (float*)(ws + 0);                  // N f32 (deg->dinv)
  float*          stats = (float*)(ws + 400128);             // 2*H f32
  unsigned short* wp1   = (unsigned short*)(ws + 402176);    // 128x256
  unsigned short* wp2   = (unsigned short*)(ws + 467712);    // 256x256
  unsigned short* wp3   = (unsigned short*)(ws + 598784);    // 256x64
  unsigned short* wp4   = (unsigned short*)(ws + 631552);    // 64x256
  unsigned short* wp5   = (unsigned short*)(ws + 664320);    // 256x256
  unsigned short* wp6   = (unsigned short*)(ws + 795392);    // 256x128
  unsigned short* bufA  = (unsigned short*)(ws + 860928);    // N*H bf16
  unsigned short* bufB  = (unsigned short*)(ws + 52060928);  // N*H bf16
  float*          bufC  = (float*)(ws + 103260928);          // N*H f32

  const int T = 256;
  const int mTiles = GN / 16;                     // 6250
  const dim3 gB((mTiles + 7) / 8);                // 782 blocks.x for GEMMs

  // degrees -> dinv
  deg_init_k<<<(GN + T - 1) / T, T, 0, stream>>>(dinv);
  deg_acc_k<<<(GE + T - 1) / T, T, 0, stream>>>(dinv, ei);
  dinv_k<<<(GN + T - 1) / T, T, 0, stream>>>(dinv);

  // pack weights into WMMA B-fragment order (bf16)
  pack_w_k<<<(GIN * GH + T - 1) / T, T, 0, stream>>>(W1, wp1, GIN, GH);
  pack_w_k<<<(GH * GH + T - 1) / T, T, 0, stream>>>(W2, wp2, GH, GH);
  pack_w_k<<<(GH * GL + T - 1) / T, T, 0, stream>>>(W3, wp3, GH, GL);
  pack_w_k<<<(GL * GH + T - 1) / T, T, 0, stream>>>(dW1, wp4, GL, GH);
  pack_w_k<<<(GH * GH + T - 1) / T, T, 0, stream>>>(dW2, wp5, GH, GH);
  pack_w_k<<<(GH * GIN + T - 1) / T, T, 0, stream>>>(dW3, wp6, GH, GIN);

  // x -> bf16
  cvt_f32_bf16_k<<<(size_t)(GN * GIN + T - 1) / T, T, 0, stream>>>(
      x, bufA, (size_t)GN * GIN);

  const size_t NH = (size_t)GN * GH;
  const size_t NL = (size_t)GN * GL;

  // ---------------- GCN layer 1 ----------------
  gemm_wmma_k<<<dim3(gB.x, GH / 16), T, 0, stream>>>(
      bufA, wp1, nullptr, bufB, nullptr, nullptr, mTiles, GIN, GH);
  agg_init_k<<<(NH + T - 1) / T, T, 0, stream>>>(bufC, bufB, b1, dinv, GH);
  agg_edges_k<<<((size_t)GE * GH / 2) / T, T, 0, stream>>>(bufC, bufB, dinv, ei, 7);
  zero_f32_k<<<2, T, 0, stream>>>(stats, 2 * GH);
  bn_stats_k<<<(GN + 63) / 64, GH, 0, stream>>>(bufC, stats, 64);
  bn_apply_k<<<(NH + T - 1) / T, T, 0, stream>>>(bufC, stats, g1, be1, bufA);

  // ---------------- GCN layer 2 ----------------
  gemm_wmma_k<<<dim3(gB.x, GH / 16), T, 0, stream>>>(
      bufA, wp2, nullptr, bufB, nullptr, nullptr, mTiles, GH, GH);
  agg_init_k<<<(NH + T - 1) / T, T, 0, stream>>>(bufC, bufB, b2, dinv, GH);
  agg_edges_k<<<((size_t)GE * GH / 2) / T, T, 0, stream>>>(bufC, bufB, dinv, ei, 7);
  zero_f32_k<<<2, T, 0, stream>>>(stats, 2 * GH);
  bn_stats_k<<<(GN + 63) / 64, GH, 0, stream>>>(bufC, stats, 64);
  bn_apply_k<<<(NH + T - 1) / T, T, 0, stream>>>(bufC, stats, g2, be2, bufA);

  // ---------------- GCN layer 3 (latent z) ----------------
  gemm_wmma_k<<<dim3(gB.x, GL / 16), T, 0, stream>>>(
      bufA, wp3, nullptr, bufB, nullptr, nullptr, mTiles, GH, GL);
  agg_init_k<<<(NL + T - 1) / T, T, 0, stream>>>(bufC, bufB, b3, dinv, GL);
  agg_edges_k<<<((size_t)GE * GL / 2) / T, T, 0, stream>>>(bufC, bufB, dinv, ei, 5);
  z_out_k<<<(NL + T - 1) / T, T, 0, stream>>>(bufC, z_f32, bufA, NL);

  // ---------------- decoder layer 1 (fused BN stats) ----------------
  zero_f32_k<<<2, T, 0, stream>>>(stats, 2 * GH);
  gemm_wmma_k<<<dim3(gB.x, GH / 16), T, 0, stream>>>(
      bufA, wp4, bufC, nullptr, db1, stats, mTiles, GL, GH);
  bn_apply_k<<<(NH + T - 1) / T, T, 0, stream>>>(bufC, stats, dg1, dbe1, bufA);

  // ---------------- decoder layer 2 (fused BN stats) ----------------
  zero_f32_k<<<2, T, 0, stream>>>(stats, 2 * GH);
  gemm_wmma_k<<<dim3(gB.x, GH / 16), T, 0, stream>>>(
      bufA, wp5, bufC, nullptr, db2, stats, mTiles, GH, GH);
  bn_apply_k<<<(NH + T - 1) / T, T, 0, stream>>>(bufC, stats, dg2, dbe2, bufA);

  // ---------------- decoder layer 3 -> x_hat ----------------
  gemm_wmma_k<<<dim3(gB.x, GIN / 16), T, 0, stream>>>(
      bufA, wp6, x_hat, nullptr, db3, nullptr, mTiles, GH, GIN);
}
